// VideoUniGraph_34462817583319
// MI455X (gfx1250) — matrix operand order repaired and checked by reading, not compiled, not run on hardware
//
#include <hip/hip_runtime.h>

// ---------------------------------------------------------------------------
// MI455X (gfx1250) implementation. All large GEMMs run on the matrix cores via
// v_wmma_f32_16x16x32_bf16 (wave32, D = A(16x32 bf16) x B(32x16 bf16) + C f32).
// fp32 activations are converted to bf16 in a bandwidth-bound pass; weights are
// repacked once per call into WMMA fragment order so B loads are one 32B load
// per lane. Edge softmax (GAT) uses float atomics with an LDS pre-reduction and
// a dedicated chunked-reduction kernel for the hot global-vertex destination.
// ---------------------------------------------------------------------------

typedef __attribute__((ext_vector_type(16))) __bf16          v16bf;
typedef __attribute__((ext_vector_type(8)))  float           v8f;
typedef __attribute__((ext_vector_type(16))) unsigned short  v16u;
typedef __attribute__((ext_vector_type(8)))  unsigned short  v8u;

__device__ __forceinline__ unsigned short f2bf(float f) {
  union { float f; unsigned u; } x; x.f = f;
  unsigned r = x.u + 0x7FFFu + ((x.u >> 16) & 1u);   // round-to-nearest-even
  return (unsigned short)(r >> 16);
}

// float atomic max via int/uint monotone trick (works for +/- and -inf init)
__device__ __forceinline__ void atomicMaxFBits(int* addr, float v) {
  if (v >= 0.f) atomicMax(addr, __float_as_int(v));
  else          atomicMin((unsigned int*)addr, __float_as_uint(v));
}
__device__ __forceinline__ void atomicMaxF(float* addr, float v) {
  atomicMaxFBits((int*)addr, v);
}

// ------------------------- bf16 convert / pack -----------------------------

__global__ __launch_bounds__(256) void cvt_f32_bf16_k(
    const float* __restrict__ in, unsigned short* __restrict__ out, size_t cnt) {
  size_t i = (size_t)blockIdx.x * blockDim.x + threadIdx.x;
  if (i < cnt) out[i] = f2bf(in[i]);
}

// Pack W (K x Nout, fp32 row-major) into WMMA B fragments:
// frag f = kt*(Nout/16)+nt ; within frag: lane-contiguous 16 bf16 per lane,
// lane = ((krem/8)%2)*16 + ncol, elem e = krem%8 + (krem/16)*8  (krem = k%32, ncol = n%16).
__global__ __launch_bounds__(256) void pack_b_bf16_k(
    const float* __restrict__ W, unsigned short* __restrict__ out, int K, int Nout) {
  int tid = blockIdx.x * blockDim.x + threadIdx.x;
  if (tid >= K * Nout) return;
  int k = tid / Nout, ncol = tid % Nout;
  int kt = k >> 5, krem = k & 31;
  int nt = ncol >> 4, nc = ncol & 15;
  int lane = (((krem >> 3) & 1) << 4) + nc;
  int e = (krem & 7) + ((krem >> 4) << 3);
  int nfrag = Nout >> 4;
  out[((size_t)(kt * nfrag + nt)) * 512 + lane * 16 + e] = f2bf(W[(size_t)k * Nout + ncol]);
}

// ------------------------------ WMMA GEMM ----------------------------------
// C[M x Nout] (fp32) (+)= A[M x K](bf16, row-major) @ Bpacked[K x Nout](bf16)
// mode 0: C = acc + bias ; mode 1: C += acc + bias
// mode 2: C += rowscale[row*rs_stride] * (acc + bias)
// Each wave computes a 16x32 slab (2 WMMAs per K-step, A fragment reused).
__global__ __launch_bounds__(256) void gemm_bf16_wmma_k(
    const unsigned short* __restrict__ A, int lda,
    const unsigned short* __restrict__ Bp,
    float* __restrict__ C, int ldc,
    const float* __restrict__ bias,
    const float* __restrict__ rowscale, int rs_stride,
    int M, int K, int Nout, int mode) {
  const int lane = threadIdx.x & 31;
  const int wave = threadIdx.x >> 5;
  const int rowtile = blockIdx.x;
  const int cg = blockIdx.y * 8 + wave;          // 32-column group
  const int ct0 = cg * 2;                        // first 16-col fragment tile
  const int nfrag = Nout >> 4;
  const int row0 = rowtile * 16;

  const int mrow  = lane & 15;
  const int khalf = lane >> 4;                   // K-half select (A layout)
  int arow = row0 + mrow; if (arow > M - 1) arow = M - 1;   // clamp tail tile
  const unsigned short* Ap  = A + (size_t)arow * lda + khalf * 8;
  const unsigned short* Bp0 = Bp + (size_t)ct0 * 512 + lane * 16;

  v8f acc0 = {0.f, 0.f, 0.f, 0.f, 0.f, 0.f, 0.f, 0.f};
  v8f acc1 = acc0;
  const int KT = K >> 5;
  for (int kt = 0; kt < KT; ++kt) {
    // A 16x32 bf16 fragment: two contiguous 8-half runs per lane
    v8u r0 = *(const v8u*)(Ap + kt * 32);
    v8u r1 = *(const v8u*)(Ap + kt * 32 + 16);
    v16u au = __builtin_shufflevector(r0, r1, 0,1,2,3,4,5,6,7,8,9,10,11,12,13,14,15);
    v16bf a = __builtin_bit_cast(v16bf, au);
    // B fragments: one 32B contiguous load per lane each
    const unsigned short* bptr = Bp0 + (size_t)kt * nfrag * 512;
    v16bf b0 = *(const v16bf*)(bptr);
    v16bf b1 = *(const v16bf*)(bptr + 512);
    acc0 = __builtin_amdgcn_wmma_f32_16x16x32_bf16(false, a, false, b0, (short)0, acc0, false, false);
    acc1 = __builtin_amdgcn_wmma_f32_16x16x32_bf16(false, a, false, b1, (short)0, acc1, false, false);
  }

  const int ccol0 = ct0 * 16 + (lane & 15);
  const int ccol1 = ccol0 + 16;
  const float bv0 = bias ? bias[ccol0] : 0.f;
  const float bv1 = bias ? bias[ccol1] : 0.f;
  const int rbase = row0 + khalf * 8;            // C layout: vgpr r -> M=r(+8)
#pragma unroll
  for (int r = 0; r < 8; ++r) {
    int crow = rbase + r;
    if (crow >= M) continue;
    size_t i0 = (size_t)crow * ldc + ccol0;
    size_t i1 = (size_t)crow * ldc + ccol1;
    float v0 = acc0[r] + bv0;
    float v1 = acc1[r] + bv1;
    if (mode == 0)      { C[i0] = v0;  C[i1] = v1; }
    else if (mode == 1) { C[i0] += v0; C[i1] += v1; }
    else {
      float g = rowscale[(size_t)crow * rs_stride];
      C[i0] += g * v0; C[i1] += g * v1;
    }
  }
}

// --------------------------- element-wise stages ---------------------------

__global__ __launch_bounds__(256) void mask_scale_k(
    float* __restrict__ x, const unsigned char* __restrict__ mask,
    const float* __restrict__ mtok, int N) {
  size_t i = (size_t)blockIdx.x * blockDim.x + threadIdx.x;
  if (i >= (size_t)N * 256) return;
  int node = (int)(i >> 8), d = (int)(i & 255);
  x[i] = mask[node] ? mtok[d] : x[i] * (1.0f / 3.0f);
}

// softmax gate + top-2 (renormalized), stored dense (zeros for unselected)
__global__ __launch_bounds__(256) void gate_topk_k(
    const float* __restrict__ x, const float* __restrict__ gw,
    const float* __restrict__ gb, float* __restrict__ g, int N) {
  int nidx = blockIdx.x * blockDim.x + threadIdx.x;
  if (nidx >= N) return;
  float logit[8];
#pragma unroll
  for (int e = 0; e < 8; ++e) logit[e] = gb[e];
  const float* xr = x + (size_t)nidx * 256;
  for (int d = 0; d < 256; ++d) {
    float xv = xr[d];
#pragma unroll
    for (int e = 0; e < 8; ++e) logit[e] += xv * gw[d * 8 + e];
  }
  int i1 = 0;
#pragma unroll
  for (int e = 1; e < 8; ++e) if (logit[e] > logit[i1]) i1 = e;
  int i2 = -1;
#pragma unroll
  for (int e = 0; e < 8; ++e)
    if (e != i1 && (i2 < 0 || logit[e] > logit[i2])) i2 = e;
  float e2 = expf(logit[i2] - logit[i1]);        // stable top-2 renorm
  float w1 = 1.f / (1.f + e2);
  float w2 = e2 * w1;
  float* gr = g + (size_t)nidx * 8;
#pragma unroll
  for (int e = 0; e < 8; ++e) gr[e] = 0.f;
  gr[i1] = w1; gr[i2] = w2;
}

__global__ __launch_bounds__(256) void init_h0_k(
    float* __restrict__ H, const float* __restrict__ gv, int n) {
  size_t i = (size_t)blockIdx.x * blockDim.x + threadIdx.x;
  if (i >= (size_t)n * 256) return;
  int node = (int)(i >> 8), d = (int)(i & 255);
  H[i] = (node == n - 1) ? gv[d] : 0.f;
}

// per-row LayerNorm + exact GELU over 256 columns (one block per row)
__global__ __launch_bounds__(256) void ln_gelu_k(
    float* __restrict__ h, const float* __restrict__ gamma,
    const float* __restrict__ beta, int N) {
  __shared__ float red[256];
  int t = threadIdx.x;
  size_t base = (size_t)blockIdx.x * 256;
  float v = h[base + t];
  red[t] = v; __syncthreads();
  for (int s = 128; s > 0; s >>= 1) { if (t < s) red[t] += red[t + s]; __syncthreads(); }
  float mu = red[0] * (1.f / 256.f); __syncthreads();
  float dv = v - mu;
  red[t] = dv * dv; __syncthreads();
  for (int s = 128; s > 0; s >>= 1) { if (t < s) red[t] += red[t + s]; __syncthreads(); }
  float var = red[0] * (1.f / 256.f);
  float xh = dv * rsqrtf(var + 1e-5f) * gamma[t] + beta[t];
  h[base + t] = 0.5f * xh * (1.f + erff(xh * 0.70710678118654752f));
}

// --------------------------------- GAT -------------------------------------

// es[i,h] = sum_d xw[i,h,d]*a_s[h,d] ; ed likewise (one block per node)
__global__ __launch_bounds__(256) void gat_scores_k(
    const float* __restrict__ xw, const float* __restrict__ as,
    const float* __restrict__ ad, float* __restrict__ es,
    float* __restrict__ ed, int n) {
  __shared__ float red[256];
  int t = threadIdx.x, node = blockIdx.x;
  const float* xr = xw + (size_t)node * 1024;
  for (int hh = 0; hh < 8; ++hh) {
    int h = hh >> 1;
    const float* a = (hh & 1) ? ad : as;
    red[t] = xr[h * 256 + t] * a[h * 256 + t];
    __syncthreads();
    for (int s = 128; s > 0; s >>= 1) { if (t < s) red[t] += red[t + s]; __syncthreads(); }
    if (t == 0) ((hh & 1) ? ed : es)[node * 4 + h] = red[0];
    __syncthreads();
  }
}

__global__ __launch_bounds__(256) void init_mz_k(float* m, float* z, int n) {
  int i = blockIdx.x * blockDim.x + threadIdx.x;
  if (i < n * 4) { m[i] = -__builtin_inff(); z[i] = 0.f; }
}

__global__ __launch_bounds__(256) void fill_bias_k(
    float* __restrict__ out, const float* __restrict__ bias, int n) {
  size_t i = (size_t)blockIdx.x * blockDim.x + threadIdx.x;
  if (i < (size_t)n * 1024) out[i] = bias[i & 1023];
}

// segment max with LDS pre-reduction for the hot global-vertex destination
__global__ __launch_bounds__(256) void edge_max_k(
    const int* __restrict__ src, const int* __restrict__ dst,
    const float* __restrict__ es, const float* __restrict__ ed,
    float* __restrict__ m, int E, int gidx) {
  __shared__ int smax[4];
  int t = threadIdx.x;
  if (t < 4) smax[t] = 0xFF800000;  // -inf bits
  __syncthreads();
  int tid = blockIdx.x * 256 + t;
  if (tid < E * 4) {
    int e = tid >> 2, h = tid & 3;
    int s = src[e], d = dst[e];
    float val = es[s * 4 + h] + ed[d * 4 + h];
    val = val > 0.f ? val : 0.2f * val;
    if (d == gidx) atomicMaxFBits(&smax[h], val);
    else           atomicMaxF(&m[(size_t)d * 4 + h], val);
  }
  __syncthreads();
  if (t < 4 && smax[t] != 0xFF800000)
    atomicMaxF(&m[(size_t)gidx * 4 + t], __int_as_float(smax[t]));
}

__global__ __launch_bounds__(256) void edge_sum_k(
    const int* __restrict__ src, const int* __restrict__ dst,
    const float* __restrict__ es, const float* __restrict__ ed,
    const float* __restrict__ m, float* __restrict__ z, int E, int gidx) {
  __shared__ float ssum[4];
  int t = threadIdx.x;
  if (t < 4) ssum[t] = 0.f;
  __syncthreads();
  int tid = blockIdx.x * 256 + t;
  if (tid < E * 4) {
    int e = tid >> 2, h = tid & 3;
    int s = src[e], d = dst[e];
    float val = es[s * 4 + h] + ed[d * 4 + h];
    val = val > 0.f ? val : 0.2f * val;
    float p = expf(val - m[(size_t)d * 4 + h]);
    if (d == gidx) atomicAdd(&ssum[h], p);
    else           atomicAdd(&z[(size_t)d * 4 + h], p);
  }
  __syncthreads();
  if (t < 4 && ssum[t] != 0.f) atomicAdd(&z[(size_t)gidx * 4 + t], ssum[t]);
}

// one wave per (edge, head); lanes cover the 256-wide feature slab.
// Edges into the global vertex are handled by gat_global_aggr_k instead.
__global__ __launch_bounds__(256) void edge_aggr_k(
    const int* __restrict__ src, const int* __restrict__ dst,
    const float* __restrict__ es, const float* __restrict__ ed,
    const float* __restrict__ m, const float* __restrict__ z,
    const float* __restrict__ xw, float* __restrict__ out, int E, int gidx) {
  int gtid = blockIdx.x * 256 + threadIdx.x;
  int w = gtid >> 5, lane = gtid & 31;
  int e = w >> 2, h = w & 3;
  if (e >= E) return;
  int s = src[e], d = dst[e];
  if (d == gidx) return;
  float val = es[s * 4 + h] + ed[d * 4 + h];
  val = val > 0.f ? val : 0.2f * val;
  float alpha = expf(val - m[(size_t)d * 4 + h]) / (z[(size_t)d * 4 + h] + 1e-16f);
  const float* xs = xw + (size_t)s * 1024 + h * 256 + lane * 8;
  float* op = out + (size_t)d * 1024 + h * 256 + lane * 8;
#pragma unroll
  for (int i = 0; i < 8; ++i) atomicAdd(&op[i], alpha * xs[i]);
}

// chunked reduction for edges whose destination is the global vertex
// grid: (HEADS, NCHUNK); each thread owns one output column of its head.
__global__ __launch_bounds__(256) void gat_global_aggr_k(
    const int* __restrict__ src, const int* __restrict__ dst,
    const float* __restrict__ es, const float* __restrict__ ed,
    const float* __restrict__ m, const float* __restrict__ z,
    const float* __restrict__ xw, float* __restrict__ out, int E, int gidx) {
  int h = blockIdx.x, t = threadIdx.x;
  int col = h * 256 + t;
  float mg = m[(size_t)gidx * 4 + h];
  float zg = z[(size_t)gidx * 4 + h] + 1e-16f;
  float edg = ed[(size_t)gidx * 4 + h];
  int per = (E + gridDim.y - 1) / gridDim.y;
  int e0 = blockIdx.y * per;
  int e1 = e0 + per; if (e1 > E) e1 = E;
  float acc = 0.f;
  for (int e = e0; e < e1; ++e) {
    if (dst[e] != gidx) continue;
    int s = src[e];
    float val = es[s * 4 + h] + edg;
    val = val > 0.f ? val : 0.2f * val;
    float alpha = expf(val - mg) / zg;
    acc += alpha * xw[(size_t)s * 1024 + col];
  }
  if (acc != 0.f) atomicAdd(&out[(size_t)gidx * 1024 + col], acc);
}

__global__ __launch_bounds__(256) void head_mean_k(
    const float* __restrict__ H, float* __restrict__ out, int n) {
  size_t i = (size_t)blockIdx.x * blockDim.x + threadIdx.x;
  if (i >= (size_t)n * 256) return;
  size_t node = i >> 8; int d = (int)(i & 255);
  const float* r = H + node * 1024 + d;
  out[i] = 0.25f * (r[0] + r[256] + r[512] + r[768]);
}

// ------------------------------ orchestration ------------------------------

extern "C" void kernel_launch(void* const* d_in, const int* in_sizes, int n_in,
                              void* d_out, int out_size, void* d_ws, size_t ws_size,
                              hipStream_t stream) {
  (void)n_in; (void)out_size; (void)ws_size;
  const int N = 65536, n = N + 1, E = in_sizes[32];

  const float* feat_v  = (const float*)d_in[0];
  const float* feat_a  = (const float*)d_in[1];
  const float* feat_t  = (const float*)d_in[2];
  const float* Wp_v    = (const float*)d_in[3];
  const float* bp_v    = (const float*)d_in[4];
  const float* Wp_a    = (const float*)d_in[5];
  const float* bp_a    = (const float*)d_in[6];
  const float* Wp_t    = (const float*)d_in[7];
  const float* bp_t    = (const float*)d_in[8];
  const float* mtok    = (const float*)d_in[9];
  const float* gate_W  = (const float*)d_in[10];
  const float* gate_b  = (const float*)d_in[11];
  const float* exp_W1  = (const float*)d_in[12];
  const float* exp_b1  = (const float*)d_in[13];
  const float* exp_g   = (const float*)d_in[14];
  const float* exp_bt  = (const float*)d_in[15];
  const float* exp_W2  = (const float*)d_in[16];
  const float* exp_b2  = (const float*)d_in[17];
  const float* gvert   = (const float*)d_in[18];
  const float* gatW[3] = {(const float*)d_in[19], (const float*)d_in[23], (const float*)d_in[27]};
  const float* gatAS[3]= {(const float*)d_in[20], (const float*)d_in[24], (const float*)d_in[28]};
  const float* gatAD[3]= {(const float*)d_in[21], (const float*)d_in[25], (const float*)d_in[29]};
  const float* gatB[3] = {(const float*)d_in[22], (const float*)d_in[26], (const float*)d_in[30]};
  const unsigned char* mask = (const unsigned char*)d_in[31];
  const int* esrc = (const int*)d_in[32];
  const int* edst = (const int*)d_in[33];
  float* outp = (float*)d_out;

  // workspace carve-out (256B aligned)
  char* w = (char*)d_ws;
  size_t off = 0;
  auto nextp = [&](size_t bytes) -> void* {
    void* p = w + off;
    off = (off + bytes + 255) & ~(size_t)255;
    return p;
  };
  unsigned short* pWv = (unsigned short*)nextp((size_t)1024 * 256 * 2);
  unsigned short* pWa = (unsigned short*)nextp((size_t)512 * 256 * 2);
  unsigned short* pWt = (unsigned short*)nextp((size_t)768 * 256 * 2);
  unsigned short* pW1 = (unsigned short*)nextp((size_t)8 * 256 * 256 * 2);
  unsigned short* pW2 = (unsigned short*)nextp((size_t)8 * 256 * 256 * 2);
  unsigned short* pG0 = (unsigned short*)nextp((size_t)256 * 1024 * 2);
  unsigned short* pG1 = (unsigned short*)nextp((size_t)1024 * 1024 * 2);
  unsigned short* pG2 = (unsigned short*)nextp((size_t)1024 * 1024 * 2);
  unsigned short* pG[3] = {pG0, pG1, pG2};
  float* gdense = (float*)nextp((size_t)N * 8 * 4);
  float* esb    = (float*)nextp((size_t)n * 4 * 4);
  float* edb    = (float*)nextp((size_t)n * 4 * 4);
  float* mb     = (float*)nextp((size_t)n * 4 * 4);
  float* zb     = (float*)nextp((size_t)n * 4 * 4);
  float* bufA   = (float*)nextp((size_t)n * 1024 * 4);   // xw / x / h_tmp
  float* bufB   = (float*)nextp((size_t)n * 1024 * 4);   // H0 / layer output
  unsigned short* bufS  = (unsigned short*)nextp((size_t)n * 1024 * 2);  // bf16 A staging
  unsigned short* bufS2 = (unsigned short*)nextp((size_t)N * 256 * 2);   // bf16 h staging

  float* xbuf = bufA;                   // N x 256
  float* htmp = bufA + (size_t)N * 256; // N x 256
  float* ybuf = bufB;                   // n x 256 (H0)

  auto packL = [&](const float* W, unsigned short* outw, int K, int Nout) {
    int total = K * Nout;
    pack_b_bf16_k<<<(total + 255) / 256, 256, 0, stream>>>(W, outw, K, Nout);
  };
  auto cvtL = [&](const float* in, unsigned short* outw, size_t cnt) {
    cvt_f32_bf16_k<<<(unsigned)((cnt + 255) / 256), 256, 0, stream>>>(in, outw, cnt);
  };
  auto gemmL = [&](const unsigned short* A, int lda, const unsigned short* Bp,
                   float* C, int ldc, const float* bias, const float* rowscale,
                   int rs_stride, int M, int K, int Nout, int mode) {
    dim3 grid((M + 15) / 16, Nout / 256);
    gemm_bf16_wmma_k<<<grid, 256, 0, stream>>>(A, lda, Bp, C, ldc, bias,
                                               rowscale, rs_stride, M, K, Nout, mode);
  };

  // ---- pack all weights to WMMA bf16 fragments
  packL(Wp_v, pWv, 1024, 256);
  packL(Wp_a, pWa, 512, 256);
  packL(Wp_t, pWt, 768, 256);
  for (int e = 0; e < 8; ++e) {
    packL(exp_W1 + (size_t)e * 65536, pW1 + (size_t)e * 65536, 256, 256);
    packL(exp_W2 + (size_t)e * 65536, pW2 + (size_t)e * 65536, 256, 256);
  }
  packL(gatW[0], pG0, 256, 1024);
  packL(gatW[1], pG1, 1024, 1024);
  packL(gatW[2], pG2, 1024, 1024);

  // ---- modal projection: x = (v@Wv+bv + a@Wa+ba + t@Wt+bt)/3, then mask
  cvtL(feat_v, bufS, (size_t)N * 1024);
  gemmL(bufS, 1024, pWv, xbuf, 256, bp_v, nullptr, 0, N, 1024, 256, 0);
  cvtL(feat_a, bufS, (size_t)N * 512);
  gemmL(bufS, 512, pWa, xbuf, 256, bp_a, nullptr, 0, N, 512, 256, 1);
  cvtL(feat_t, bufS, (size_t)N * 768);
  gemmL(bufS, 768, pWt, xbuf, 256, bp_t, nullptr, 0, N, 768, 256, 1);
  mask_scale_k<<<(unsigned)(((size_t)N * 256 + 255) / 256), 256, 0, stream>>>(xbuf, mask, mtok, N);

  // ---- MoE: top-2 gate, dense all-expert compute with scaled accumulation
  gate_topk_k<<<(N + 255) / 256, 256, 0, stream>>>(xbuf, gate_W, gate_b, gdense, N);
  init_h0_k<<<(unsigned)(((size_t)n * 256 + 255) / 256), 256, 0, stream>>>(ybuf, gvert, n);
  cvtL(xbuf, bufS, (size_t)N * 256);            // x in bf16, reused by all experts
  for (int e = 0; e < 8; ++e) {
    gemmL(bufS, 256, pW1 + (size_t)e * 65536, htmp, 256, exp_b1 + e * 256,
          nullptr, 0, N, 256, 256, 0);
    ln_gelu_k<<<N, 256, 0, stream>>>(htmp, exp_g + e * 256, exp_bt + e * 256, N);
    cvtL(htmp, bufS2, (size_t)N * 256);
    gemmL(bufS2, 256, pW2 + (size_t)e * 65536, ybuf, 256, exp_b2 + e * 256,
          gdense + e, 8, N, 256, 256, 2);       // y += g_e * (h@W2 + b2)
  }

  // ---- 3 GAT layers (ping-pong bufB -> bufA(xw) -> bufB)
  const int Din[3] = {256, 1024, 1024};
  float* Hin = ybuf;
  for (int l = 0; l < 3; ++l) {
    cvtL(Hin, bufS, (size_t)n * Din[l]);
    gemmL(bufS, Din[l], pG[l], bufA, 1024, nullptr, nullptr, 0, n, Din[l], 1024, 0);
    gat_scores_k<<<n, 256, 0, stream>>>(bufA, gatAS[l], gatAD[l], esb, edb, n);
    init_mz_k<<<(n * 4 + 255) / 256, 256, 0, stream>>>(mb, zb, n);
    fill_bias_k<<<(unsigned)(((size_t)n * 1024 + 255) / 256), 256, 0, stream>>>(bufB, gatB[l], n);
    edge_max_k<<<(E * 4 + 255) / 256, 256, 0, stream>>>(esrc, edst, esb, edb, mb, E, N);
    edge_sum_k<<<(E * 4 + 255) / 256, 256, 0, stream>>>(esrc, edst, esb, edb, mb, zb, E, N);
    edge_aggr_k<<<(unsigned)(((size_t)E * 128 + 255) / 256), 256, 0, stream>>>(
        esrc, edst, esb, edb, mb, zb, bufA, bufB, E, N);
    dim3 gg(4, 128);
    gat_global_aggr_k<<<gg, 256, 0, stream>>>(esrc, edst, esb, edb, mb, zb, bufA, bufB, E, N);
    Hin = bufB;
  }

  // ---- mean over heads -> (n, 256)
  head_mean_k<<<(unsigned)(((size_t)n * 256 + 255) / 256), 256, 0, stream>>>(bufB, outp, n);
}
